// RNN_6296422056099
// MI455X (gfx1250) — compile-verified, matrix operand order checked by reading, not run on hardware
//
#include <hip/hip_runtime.h>

// ---------------------------------------------------------------------------
// RNN on MI455X (gfx1250, wave32, WMMA bf16 16x16x32)
//   prep : W_ih -> bf16 [64][224] (zero-padded), W_hh/W1 -> bf16
//   proj : xp tiles (16 batch x 64 H at fixed t) via WMMA, stored as packed
//          bf16 C-fragments (one b128 store per lane per N-tile)
//   scan : h_t = tanh(xp_t + h_{t-1} @ W_hh^T); 8 blocks x 4 waves,
//          LDS double-buffered transpose, h_t dumped as raw A-fragments
//   head : softmax(relu(hs @ W1^T + b1) @ W2^T + b2), A-frags read directly
// ---------------------------------------------------------------------------

typedef __attribute__((ext_vector_type(16))) __bf16 v16bf;
typedef __attribute__((ext_vector_type(8)))  __bf16 v8bf;
typedef __attribute__((ext_vector_type(8)))  float  v8f;
typedef __attribute__((ext_vector_type(4)))  float  v4f;

#define B_ 128
#define T_ 2048
#define I_ 200
#define IP_ 224   // I padded to multiple of 32
#define H_ 64

#define WMMA_BF16(A, BM, C) \
  __builtin_amdgcn_wmma_f32_16x16x32_bf16(false, (A), false, (BM), (short)0, (C), false, false)

__device__ __forceinline__ v16bf cat8(v8bf lo, v8bf hi) {
  return __builtin_shufflevector(lo, hi, 0, 1, 2, 3, 4, 5, 6, 7, 8, 9, 10, 11, 12, 13, 14, 15);
}

__device__ __forceinline__ v16bf bf_zero() {
  v16bf r;
#pragma unroll
  for (int i = 0; i < 16; ++i) r[i] = (__bf16)0.0f;
  return r;
}

// bf16 A/B fragment from a bf16 row: two b128 loads, zero conversion.
// ISA 16-bit A/B layout: lanes 0-15 k = kbase+{0..7,16..23}, lanes 16-31 +8.
__device__ __forceinline__ v16bf load_fragbf(const __bf16* __restrict__ p,
                                             int kbase, int hsel) {
  const v8bf* q0 = (const v8bf*)(p + kbase + hsel * 8);
  const v8bf* q1 = (const v8bf*)(p + kbase + 16 + hsel * 8);
  return cat8(*q0, *q1);
}

// f32 source fragment (x stream): 4x b128 loads + pk converts, K-tail guarded.
__device__ __forceinline__ v16bf gather16_f32(const float* __restrict__ p,
                                              int kbase, int hsel, int limit) {
  v16bf r;
  if (kbase + 32 <= limit) {
    const float4* q0 = (const float4*)(p + kbase + hsel * 8);
    const float4* q1 = (const float4*)(p + kbase + 16 + hsel * 8);
    float4 f0 = q0[0], f1 = q0[1], f2 = q1[0], f3 = q1[1];
    r[0] = (__bf16)f0.x;  r[1] = (__bf16)f0.y;  r[2] = (__bf16)f0.z;  r[3] = (__bf16)f0.w;
    r[4] = (__bf16)f1.x;  r[5] = (__bf16)f1.y;  r[6] = (__bf16)f1.z;  r[7] = (__bf16)f1.w;
    r[8] = (__bf16)f2.x;  r[9] = (__bf16)f2.y;  r[10] = (__bf16)f2.z; r[11] = (__bf16)f2.w;
    r[12] = (__bf16)f3.x; r[13] = (__bf16)f3.y; r[14] = (__bf16)f3.z; r[15] = (__bf16)f3.w;
  } else {
    const int k0 = kbase + hsel * 8, k1 = kbase + 16 + hsel * 8;
#pragma unroll
    for (int i = 0; i < 8; ++i) { int k = k0 + i; r[i] = (__bf16)((k < limit) ? p[k] : 0.0f); }
#pragma unroll
    for (int i = 0; i < 8; ++i) { int k = k1 + i; r[8 + i] = (__bf16)((k < limit) ? p[k] : 0.0f); }
  }
  return r;
}

__device__ __forceinline__ unsigned pack2_bf16(float a, float b) {
  unsigned short ua = __builtin_bit_cast(unsigned short, (__bf16)a);
  unsigned short ub = __builtin_bit_cast(unsigned short, (__bf16)b);
  return ((unsigned)ub << 16) | (unsigned)ua;
}

__device__ __forceinline__ float tanh_fast(float x) {
  float cx = fminf(fmaxf(x, -9.0f), 9.0f);
  float e  = __builtin_amdgcn_exp2f(cx * 2.8853900817779268f);  // 2*log2(e)
  return (e - 1.0f) * __builtin_amdgcn_rcpf(e + 1.0f);
}

// ---------------------------------------------------------------------------
// Prep: weights -> bf16 once (W_ih zero-padded to [64][224]).
// ---------------------------------------------------------------------------
__global__ void __launch_bounds__(256)
rnn_prep_kernel(const float* __restrict__ Wih, const float* __restrict__ Whh,
                const float* __restrict__ W1, __bf16* __restrict__ wihb,
                __bf16* __restrict__ whhb, __bf16* __restrict__ w1b) {
  for (int i = threadIdx.x; i < 64 * IP_; i += 256) {
    int r = i / IP_, k = i - r * IP_;
    wihb[i] = (__bf16)((k < I_) ? Wih[r * I_ + k] : 0.0f);
  }
  for (int i = threadIdx.x; i < 64 * 64; i += 256) whhb[i] = (__bf16)Whh[i];
  for (int i = threadIdx.x; i < 16 * 64; i += 256) w1b[i] = (__bf16)W1[i];
}

// ---------------------------------------------------------------------------
// Proj: tile u = t*8 + btile -> [16 batch x 64 H] at fixed t.  7 K-chunks,
// 4 N-tiles -> 28 WMMAs/wave.  Output = packed bf16 C-fragment dump:
// dword i of lane = rows (2i, 2i+1) -> exactly what the scan's C init wants.
// ---------------------------------------------------------------------------
__global__ void __launch_bounds__(256)
rnn_proj_kernel(const float* __restrict__ x, const __bf16* __restrict__ wihb,
                const float* __restrict__ bih, const float* __restrict__ bhh,
                unsigned* __restrict__ xp32) {
  const int lane = threadIdx.x & 31;
  const int wave = threadIdx.x >> 5;
  const int hsel = lane >> 4;
  const int col  = lane & 15;
  const size_t u = (size_t)blockIdx.x * 8 + wave;  // 0..16383
  const int t     = (int)(u >> 3);
  const int btile = (int)(u & 7);

  v8f acc[4];
#pragma unroll
  for (int n = 0; n < 4; ++n) {
    float bias = bih[n * 16 + col] + bhh[n * 16 + col];  // C layout: lane == col
#pragma unroll
    for (int r = 0; r < 8; ++r) acc[n][r] = bias;
  }

  const float* xrow = x + ((size_t)(btile * 16 + col) * T_ + t) * I_;
#pragma unroll
  for (int k = 0; k < 7; ++k) {
    v16bf a = gather16_f32(xrow, k * 32, hsel, I_);
#pragma unroll
    for (int n = 0; n < 4; ++n) {
      v16bf b = load_fragbf(wihb + (size_t)(n * 16 + col) * IP_, k * 32, hsel);
      acc[n] = WMMA_BF16(a, b, acc[n]);
    }
  }

#pragma unroll
  for (int n = 0; n < 4; ++n) {
    uint4 pk = make_uint4(pack2_bf16(acc[n][0], acc[n][1]), pack2_bf16(acc[n][2], acc[n][3]),
                          pack2_bf16(acc[n][4], acc[n][5]), pack2_bf16(acc[n][6], acc[n][7]));
    *(uint4*)(xp32 + (u * 4 + n) * 128 + lane * 4) = pk;
  }
}

// ---------------------------------------------------------------------------
// Scan: 8 blocks (16 batch rows) x 4 waves (one 16-wide N-tile each).
// Per step: 1 b128 xp load (prefetched) -> 2 WMMA (K=64) -> tanh ->
// LDS transpose (stride 72, conflict-free b128 reads) -> 1 barrier ->
// dump h_t A-fragment quarter to global (1 b128 store/wave).
// ---------------------------------------------------------------------------
__global__ void __launch_bounds__(128)
rnn_scan_kernel(const unsigned* __restrict__ xp32, const __bf16* __restrict__ whhb,
                __bf16* __restrict__ hsb) {
  __shared__ __align__(16) __bf16 hbuf[2][16 * 72];

  const int lane  = threadIdx.x & 31;
  const int ntile = threadIdx.x >> 5;  // 0..3
  const int hsel  = lane >> 4;
  const int col   = lane & 15;
  const int btile = blockIdx.x;

  const __bf16* wrow = whhb + (size_t)(ntile * 16 + col) * H_;  // B col n = W_hh row n
  v16bf bw0 = load_fragbf(wrow, 0, hsel);
  v16bf bw1 = load_fragbf(wrow, 32, hsel);

  v16bf a0 = bf_zero(), a1 = bf_zero();  // h_{-1} = 0

  const unsigned* xpw = xp32 + ((size_t)btile * 4 + ntile) * 128 + lane * 4;
  const size_t xstep = 8 * 4 * 128;  // dwords per timestep

  uint4 cur = *(const uint4*)(xpw);

  for (int t = 0; t < T_; ++t) {
    v8f acc;
    acc[0] = __uint_as_float(cur.x << 16); acc[1] = __uint_as_float(cur.x & 0xffff0000u);
    acc[2] = __uint_as_float(cur.y << 16); acc[3] = __uint_as_float(cur.y & 0xffff0000u);
    acc[4] = __uint_as_float(cur.z << 16); acc[5] = __uint_as_float(cur.z & 0xffff0000u);
    acc[6] = __uint_as_float(cur.w << 16); acc[7] = __uint_as_float(cur.w & 0xffff0000u);
    acc = WMMA_BF16(a0, bw0, acc);
    acc = WMMA_BF16(a1, bw1, acc);

    uint4 nxt = make_uint4(0u, 0u, 0u, 0u);
    if (t + 1 < T_) {
      nxt = *(const uint4*)(xpw + (size_t)(t + 1) * xstep);
      __builtin_prefetch(xpw + (size_t)(t + 2 < T_ ? t + 2 : t + 1) * xstep, 0, 0);
    }

    __bf16* buf = hbuf[t & 1];
#pragma unroll
    for (int r = 0; r < 8; ++r)
      buf[(r + hsel * 8) * 72 + ntile * 16 + col] = (__bf16)tanh_fast(acc[r]);
    __syncthreads();  // publish h_t (prior buffer reads consumed pre-store)

    const __bf16* rb = buf + col * 72 + hsel * 8;
    a0 = cat8(*(const v8bf*)(rb), *(const v8bf*)(rb + 16));
    a1 = cat8(*(const v8bf*)(rb + 32), *(const v8bf*)(rb + 48));

    // dump this wave's quarter of the h_t A-fragment (head reads it verbatim)
    v8bf part;
    if      (ntile == 0) part = __builtin_shufflevector(a0, a0, 0, 1, 2, 3, 4, 5, 6, 7);
    else if (ntile == 1) part = __builtin_shufflevector(a0, a0, 8, 9, 10, 11, 12, 13, 14, 15);
    else if (ntile == 2) part = __builtin_shufflevector(a1, a1, 0, 1, 2, 3, 4, 5, 6, 7);
    else                 part = __builtin_shufflevector(a1, a1, 8, 9, 10, 11, 12, 13, 14, 15);
    *(v8bf*)(hsb + ((size_t)(t * 8 + btile) * 4 + ntile) * 256 + lane * 8) = part;

    cur = nxt;
  }
}

// ---------------------------------------------------------------------------
// Head: one wave per 2 tiles (u, u+1); A-fragments loaded verbatim from the
// scan's dump (4 b128 loads/tile), relu(WMMA+b1) staged to LDS (stride 17),
// then each lane owns one row: 16->4 matmul + softmax, b128 store.
// ---------------------------------------------------------------------------
__global__ void __launch_bounds__(256)
rnn_head_kernel(const __bf16* __restrict__ hsb, const __bf16* __restrict__ w1b,
                const float* __restrict__ b1, const float* __restrict__ W2,
                const float* __restrict__ b2, float* __restrict__ out) {
  __shared__ __align__(16) float red[8][32 * 17];

  const int lane = threadIdx.x & 31;
  const int wave = threadIdx.x >> 5;
  const int hsel = lane >> 4;
  const int col  = lane & 15;
  const size_t u0 = ((size_t)blockIdx.x * 8 + wave) * 2;

  v16bf bw0 = load_fragbf(w1b + (size_t)col * H_, 0, hsel);
  v16bf bw1 = load_fragbf(w1b + (size_t)col * H_, 32, hsel);
  const float bias = b1[col];
  float* my = red[wave];

#pragma unroll
  for (int sub = 0; sub < 2; ++sub) {
    const __bf16* fb = hsb + (u0 + sub) * 1024 + lane * 8;
    v16bf a0 = cat8(*(const v8bf*)(fb),       *(const v8bf*)(fb + 256));
    v16bf a1 = cat8(*(const v8bf*)(fb + 512), *(const v8bf*)(fb + 768));
    v8f acc;
#pragma unroll
    for (int r = 0; r < 8; ++r) acc[r] = bias;
    acc = WMMA_BF16(a0, bw0, acc);
    acc = WMMA_BF16(a1, bw1, acc);
#pragma unroll
    for (int r = 0; r < 8; ++r)
      my[(sub * 16 + r + hsel * 8) * 17 + col] = fmaxf(acc[r], 0.0f);
  }

  // per-wave LDS region; DS ops are in-order within a wave -> no barrier
  const int row = lane;  // 32 staged rows, one per lane
  float v[16];
#pragma unroll
  for (int k = 0; k < 16; ++k) v[k] = my[row * 17 + k];
  float o[4];
#pragma unroll
  for (int j = 0; j < 4; ++j) {
    float s = b2[j];
#pragma unroll
    for (int k = 0; k < 16; ++k) s = fmaf(v[k], W2[j * 16 + k], s);
    o[j] = s;
  }
  float m = fmaxf(fmaxf(o[0], o[1]), fmaxf(o[2], o[3]));
  float e[4], ssum = 0.0f;
#pragma unroll
  for (int j = 0; j < 4; ++j) {
    e[j] = __builtin_amdgcn_exp2f((o[j] - m) * 1.4426950408889634f);
    ssum += e[j];
  }
  float inv = __builtin_amdgcn_rcpf(ssum);
  v4f res = {e[0] * inv, e[1] * inv, e[2] * inv, e[3] * inv};

  // row -> (b, t):  u = u0 + (lane>>4); tile = (t*8 + btile)
  const size_t u = u0 + (lane >> 4);
  const size_t t = u >> 3;
  const size_t b = (u & 7) * 16 + (lane & 15);
  *(v4f*)(out + (b * T_ + t) * 4) = res;
}

// ---------------------------------------------------------------------------
extern "C" void kernel_launch(void* const* d_in, const int* in_sizes, int n_in,
                              void* d_out, int out_size, void* d_ws, size_t ws_size,
                              hipStream_t stream) {
  const float* x   = (const float*)d_in[0];
  const float* Wih = (const float*)d_in[1];
  const float* Whh = (const float*)d_in[2];
  const float* bih = (const float*)d_in[3];
  const float* bhh = (const float*)d_in[4];
  const float* W1  = (const float*)d_in[5];
  const float* b1  = (const float*)d_in[6];
  const float* W2  = (const float*)d_in[7];
  const float* b2  = (const float*)d_in[8];
  float* out = (float*)d_out;

  // workspace: xp packed bf16 frags (33.5 MB) | hs A-frags bf16 (33.5 MB) | bf16 weights (~39 KB)
  const size_t xp_bytes = (size_t)B_ * T_ * H_ * 2;
  unsigned* xp32 = (unsigned*)d_ws;
  __bf16* hsb  = (__bf16*)((char*)d_ws + xp_bytes);
  __bf16* wihb = (__bf16*)((char*)d_ws + 2 * xp_bytes);
  __bf16* whhb = wihb + 64 * IP_;
  __bf16* w1b  = whhb + 64 * 64;

  rnn_prep_kernel<<<1, 256, 0, stream>>>(Wih, Whh, W1, wihb, whhb, w1b);
  rnn_proj_kernel<<<2048, 256, 0, stream>>>(x, wihb, bih, bhh, xp32);
  rnn_scan_kernel<<<8, 128, 0, stream>>>(xp32, whhb, hsb);
  rnn_head_kernel<<<1024, 256, 0, stream>>>(hsb, w1b, b1, W2, b2, out);
}
